// SwinTransformer_71021579207118
// MI455X (gfx1250) — compile-verified, matrix operand order checked by reading, not run on hardware
//
#include <hip/hip_runtime.h>
#include <math.h>

// ---------------------------------------------------------------------------
// Swin block for MI455X (gfx1250, wave32, WMMA).
// One workgroup = one 7x7 shifted window (padded 49 -> 64 rows).
// All GEMMs via v_wmma_f32_16x16x32_bf16 (bf16 in, f32 accumulate).
// Weights converted once to bf16 in d_ws (221 KB -> lives in 192MB L2).
// ---------------------------------------------------------------------------

typedef __bf16 bf16_t;
typedef __attribute__((ext_vector_type(16))) __bf16 v16bf;
typedef __attribute__((ext_vector_type(8)))  float  v8f;
typedef __attribute__((ext_vector_type(4)))  unsigned int uv4;

#define CDIM 96
#define FFD  384
#define LW   49      // window length
#define LP   64      // padded window length
#define EPS  1e-5f

// ---- LDS layout (byte offsets). FFN buffer aliases attention buffers. ----
#define OFF_X    0        // float  [64*96]   xw / x2 residual
#define OFF_H    24576    // bf16   [64*96]   h (LN1) then h2 (LN2)
#define OFF_O    36864    // bf16   [64*96]   attention output
#define OFF_QK   49152    // bf16   [64*192]  q | k (row-major, q pre-scaled)
#define OFF_VT   73728    // bf16   [3][32][64] v^T per head
#define OFF_SC   86016    // float  [3][64][64] scores; bf16 probs in-place
#define OFF_T    49152    // bf16   [64*384]  FFN mid  (aliases QK/VT/SC)
#define OFF_DST  135168   // int    [64]      per-token scatter/gather offset
#define OFF_RID  135424   // int    [64]      per-token swin-mask region id
#define LDS_BYTES 135680

// Fragment loader: 16 bf16 per lane as two aligned b128 chunks.
// Lane l (g=l>>4, i=l&15) covers row (r0+i), K slots {k0+8g..+7, k0+16+8g..+7},
// matching the documented 16-bit 16x32 A-operand striping. B-operands are
// fetched with the same pattern from the "transposed source" (weight rows /
// token rows / vT rows), so each WMMA computes C = A_rows x (B_rows)^T.
__device__ __forceinline__ v16bf load_frag(const bf16_t* base, int stride,
                                           int r0, int k0, int lane) {
  int g = lane >> 4, i = lane & 15;
  const bf16_t* p = base + (size_t)(r0 + i) * stride + k0 + 8 * g;
  union { uv4 q[2]; v16bf v; } u;
  u.q[0] = *(const uv4*)(p);
  u.q[1] = *(const uv4*)(p + 16);
  return u.v;
}

__device__ __forceinline__ v8f wmma_bf16(v16bf a, v16bf b, v8f c) {
  return __builtin_amdgcn_wmma_f32_16x16x32_bf16(false, a, false, b,
                                                 (short)0, c, false, false);
}

// LayerNorm of a 64x96 f32 LDS block -> bf16 LDS block. 4 lanes per row.
__device__ __forceinline__ void ln_to_bf16(const float* src, bf16_t* dst,
                                           const float* wgt, const float* bia,
                                           int t) {
  int row = t >> 2, sub = t & 3;
  const float* p = src + row * CDIM;
  float s = 0.f, ss = 0.f;
  for (int j = 0; j < 24; ++j) { float v = p[sub * 24 + j]; s += v; ss += v * v; }
  s  += __shfl_xor(s, 1, 32);   ss += __shfl_xor(ss, 1, 32);
  s  += __shfl_xor(s, 2, 32);   ss += __shfl_xor(ss, 2, 32);
  float mu  = s * (1.f / 96.f);
  float var = ss * (1.f / 96.f) - mu * mu;
  float inv = rsqrtf(var + EPS);
  bf16_t* q = dst + row * CDIM;
  if (row < LW) {
    for (int j = 0; j < 24; ++j) {
      int c = sub * 24 + j;
      q[c] = (bf16_t)((p[c] - mu) * inv * wgt[c] + bia[c]);
    }
  } else {
    for (int j = 0; j < 24; ++j) q[sub * 24 + j] = (bf16_t)0.f;
  }
}

__global__ void cvt_bf16(const float* __restrict__ src, bf16_t* __restrict__ dst, int n) {
  int i = blockIdx.x * 256 + threadIdx.x;
  if (i < n) dst[i] = (bf16_t)src[i];
}

__global__ __launch_bounds__(256) void swin_block_kernel(
    const float* __restrict__ x,
    const bf16_t* __restrict__ qkv_w, const float* __restrict__ qkv_b,
    const bf16_t* __restrict__ out_w, const float* __restrict__ out_b,
    const float* __restrict__ ln1_w,  const float* __restrict__ ln1_b,
    const float* __restrict__ ln2_w,  const float* __restrict__ ln2_b,
    const bf16_t* __restrict__ ff1_w, const float* __restrict__ ff1_b,
    const bf16_t* __restrict__ ff2_w, const float* __restrict__ ff2_b,
    float* __restrict__ out) {
  extern __shared__ unsigned char smem[];
  float*  s_x   = (float*)(smem + OFF_X);
  bf16_t* s_h   = (bf16_t*)(smem + OFF_H);
  bf16_t* s_o   = (bf16_t*)(smem + OFF_O);
  bf16_t* s_qk  = (bf16_t*)(smem + OFF_QK);
  bf16_t* s_vt  = (bf16_t*)(smem + OFF_VT);
  float*  s_sc  = (float*)(smem + OFF_SC);
  bf16_t* s_t   = (bf16_t*)(smem + OFF_T);
  int*    s_dst = (int*)(smem + OFF_DST);
  int*    s_rid = (int*)(smem + OFF_RID);

  const int t    = threadIdx.x;
  // readfirstlane => compiler-provable wave-uniform: all tile-loop induction,
  // tile coordinates and base addresses go to SGPRs / scalar branches.
  const int wave = __builtin_amdgcn_readfirstlane(t >> 5);
  const int lane = t & 31;
  const int lo   = lane & 15, lg = lane >> 4;
  const int b    = blockIdx.x >> 6;
  const int w    = blockIdx.x & 63;
  const int wy   = w >> 3, wx = w & 7;
  const size_t xbase = (size_t)b * 3136;

  // ---- phase 0a: per-token tables (shift mapping + swin-mask region id) ----
  if (t < LP) {
    if (t < LW) {
      int ty = t / 7, tx = t % 7;
      int gy = wy * 7 + ty, gx = wx * 7 + tx;
      // cyclic shift roll(-3): src/dst image coord = (g+3)%56
      s_dst[t] = ((gy + 3) % 56) * 56 + ((gx + 3) % 56);
      int ry = (gy < 49) ? 0 : ((gy < 53) ? 1 : 2);
      int rx = (gx < 49) ? 0 : ((gx < 53) ? 1 : 2);
      s_rid[t] = ry * 3 + rx;
    } else {
      s_dst[t] = -1;
      s_rid[t] = -1;
    }
  }
  __syncthreads();

  // ---- phase 0b: gather window (coalesced, table-driven) ----
  for (int idx = t; idx < LP * CDIM; idx += 256) {
    int m = idx / CDIM, c = idx - m * CDIM;
    s_x[idx] = (m < LW) ? x[(xbase + s_dst[m]) * CDIM + c] : 0.f;
  }
  __syncthreads();
  ln_to_bf16(s_x, s_h, ln1_w, ln1_b, t);
  __syncthreads();

  // ---- phase 1: QKV = h @ qkv_w^T + b  (64x288). Three branch-free loops. ----
  // 1q: q tiles (n0 in [0,96)), pre-scaled by 1/sqrt(32)
  for (int tile = wave; tile < 24; tile += 8) {
    int m0 = (tile / 6) << 4, n0 = (tile % 6) << 4;
    v8f acc = {};
    for (int k0 = 0; k0 < CDIM; k0 += 32) {
      v16bf a  = load_frag(s_h,   CDIM, m0, k0, lane);
      v16bf bm = load_frag(qkv_w, CDIM, n0, k0, lane);
      acc = wmma_bf16(a, bm, acc);
    }
    int n = n0 + lo;
    float bias = qkv_b[n];
    for (int r = 0; r < 8; ++r)
      s_qk[(m0 + r + 8 * lg) * 192 + n] =
          (bf16_t)((acc[r] + bias) * 0.17677669529663687f);
  }
  // 1k: k tiles (n0 in [96,192))
  for (int tile = wave; tile < 24; tile += 8) {
    int m0 = (tile / 6) << 4, n0 = 96 + ((tile % 6) << 4);
    v8f acc = {};
    for (int k0 = 0; k0 < CDIM; k0 += 32) {
      v16bf a  = load_frag(s_h,   CDIM, m0, k0, lane);
      v16bf bm = load_frag(qkv_w, CDIM, n0, k0, lane);
      acc = wmma_bf16(a, bm, acc);
    }
    int n = n0 + lo;
    float bias = qkv_b[n];
    for (int r = 0; r < 8; ++r)
      s_qk[(m0 + r + 8 * lg) * 192 + n] = (bf16_t)(acc[r] + bias);
  }
  // 1v: v tiles (n0 in [192,288)), stored transposed; 8 rows contiguous -> b128
  for (int tile = wave; tile < 24; tile += 8) {
    int m0 = (tile / 6) << 4, nv = (tile % 6) << 4;   // nv in [0,96)
    v8f acc = {};
    for (int k0 = 0; k0 < CDIM; k0 += 32) {
      v16bf a  = load_frag(s_h,   CDIM, m0, k0, lane);
      v16bf bm = load_frag(qkv_w, CDIM,192 + nv, k0, lane);
      acc = wmma_bf16(a, bm, acc);
    }
    int nn = nv + lo;
    float bias = qkv_b[192 + nn];
    bf16_t* vrow = s_vt + (nn >> 5) * 2048 + (nn & 31) * 64 + m0 + 8 * lg;
    union { uv4 q; bf16_t e[8]; } pk;
    for (int r = 0; r < 8; ++r) pk.e[r] = (bf16_t)(acc[r] + bias);
    *(uv4*)vrow = pk.q;
  }
  __syncthreads();

  // ---- phase 2a: scores = q @ k^T + swin mask (+pad mask), 3 heads ----
  for (int tile = wave; tile < 48; tile += 8) {
    int h = tile >> 4, r2 = tile & 15;
    int m0 = (r2 >> 2) << 4, n0 = (r2 & 3) << 4;
    v16bf a  = load_frag(s_qk + h * 32,      192, m0, 0, lane);
    v16bf bm = load_frag(s_qk + 96 + h * 32, 192, n0, 0, lane);
    v8f acc = {};
    acc = wmma_bf16(a, bm, acc);
    int n = n0 + lo;
    int ridc = (n < LW) ? s_rid[n] : -2;   // -2 never matches s_rid (>=-1)
    float* dst = s_sc + h * 4096;
    for (int r = 0; r < 8; ++r) {
      int m = m0 + r + 8 * lg;
      float v = acc[r];
      if (n >= LW) v = -1e9f;
      else if (s_rid[m] != ridc) v += -1e9f;   // rows >= LW poisoned; unused
      dst[m * 64 + n] = v;
    }
  }
  __syncthreads();

  // ---- phase 2b: softmax rows, write bf16 probs in-place ----
  if (t < 192) {
    int h = t >> 6, m = t & 63;
    float*  row  = s_sc + h * 4096 + m * 64;
    bf16_t* prow = (bf16_t*)row;
    if (m < LW) {
      float mx = -1e30f;
      for (int c = 0; c < LW; ++c) mx = fmaxf(mx, row[c]);
      float sum = 0.f;
      for (int c = 0; c < LW; ++c) {        // safe: prow[c] clobbers row[c>>1] only
        float e = __expf(row[c] - mx);
        sum += e;
        prow[c] = (bf16_t)e;
      }
      float inv = 1.f / sum;
      for (int c = 0; c < LW; ++c) prow[c] = (bf16_t)((float)prow[c] * inv);
      for (int c = LW; c < 64; ++c) prow[c] = (bf16_t)0.f;
    } else {
      for (int c = 0; c < 64; ++c) prow[c] = (bf16_t)0.f;
    }
  }
  __syncthreads();

  // ---- phase 2c: O_h = attn_h @ v_h (via v^T rows) ----
  for (int tile = wave; tile < 24; tile += 8) {
    int h = tile >> 3, r2 = tile & 7;
    int m0 = (r2 >> 1) << 4, n0 = (r2 & 1) << 4;
    const bf16_t* attn = (const bf16_t*)(s_sc + h * 4096);
    v8f acc = {};
    for (int k0 = 0; k0 < 64; k0 += 32) {
      v16bf a  = load_frag(attn,            128, m0, k0, lane);
      v16bf bm = load_frag(s_vt + h * 2048,  64, n0, k0, lane);
      acc = wmma_bf16(a, bm, acc);
    }
    int n = n0 + lo;
    for (int r = 0; r < 8; ++r)
      s_o[(m0 + r + 8 * lg) * CDIM + h * 32 + n] = (bf16_t)acc[r];
  }
  __syncthreads();

  // ---- phase 3: x2 = xw + O @ out_w^T + out_b (residual in-place in s_x) ----
  for (int tile = wave; tile < 24; tile += 8) {
    int m0 = (tile / 6) << 4, n0 = (tile % 6) << 4;
    v8f acc = {};
    for (int k0 = 0; k0 < CDIM; k0 += 32) {
      v16bf a  = load_frag(s_o,   CDIM, m0, k0, lane);
      v16bf bm = load_frag(out_w, CDIM, n0, k0, lane);
      acc = wmma_bf16(a, bm, acc);
    }
    int n = n0 + lo;
    float bias = out_b[n];
    for (int r = 0; r < 8; ++r) {
      int m = m0 + r + 8 * lg;
      s_x[m * CDIM + n] += acc[r] + bias;
    }
  }
  __syncthreads();
  ln_to_bf16(s_x, s_h, ln2_w, ln2_b, t);   // h2
  __syncthreads();

  // ---- phase 4: t = gelu(h2 @ ff1_w^T + b) (exact erf GELU), 64x384 ----
  for (int tile = wave; tile < 96; tile += 8) {
    int m0 = (tile / 24) << 4, n0 = (tile % 24) << 4;
    v8f acc = {};
    for (int k0 = 0; k0 < CDIM; k0 += 32) {
      v16bf a  = load_frag(s_h,   CDIM, m0, k0, lane);
      v16bf bm = load_frag(ff1_w, CDIM, n0, k0, lane);
      acc = wmma_bf16(a, bm, acc);
    }
    int n = n0 + lo;
    float bias = ff1_b[n];
    for (int r = 0; r < 8; ++r) {
      int m = m0 + r + 8 * lg;
      float v = acc[r] + bias;
      v = 0.5f * v * (1.f + erff(v * 0.70710678118654752f));
      s_t[m * FFD + n] = (bf16_t)v;
    }
  }
  __syncthreads();

  // ---- phase 5: out = x2 + t @ ff2_w^T + b, scatter with reverse shift ----
  for (int tile = wave; tile < 24; tile += 8) {
    int m0 = (tile / 6) << 4, n0 = (tile % 6) << 4;
    v8f acc = {};
    for (int k0 = 0; k0 < FFD; k0 += 32) {
      v16bf a  = load_frag(s_t,   FFD, m0, k0, lane);
      v16bf bm = load_frag(ff2_w, FFD, n0, k0, lane);
      acc = wmma_bf16(a, bm, acc);
    }
    int n = n0 + lo;
    float bias = ff2_b[n];
    for (int r = 0; r < 8; ++r) {
      int m = m0 + r + 8 * lg;
      int d = s_dst[m];
      if (d >= 0)
        out[(xbase + d) * CDIM + n] = s_x[m * CDIM + n] + acc[r] + bias;
    }
  }
}

extern "C" void kernel_launch(void* const* d_in, const int* in_sizes, int n_in,
                              void* d_out, int out_size, void* d_ws, size_t ws_size,
                              hipStream_t stream) {
  (void)in_sizes; (void)n_in; (void)out_size; (void)ws_size;
  const float* x     = (const float*)d_in[0];
  const float* qkv_w = (const float*)d_in[1];
  const float* qkv_b = (const float*)d_in[2];
  const float* out_w = (const float*)d_in[3];
  const float* out_b = (const float*)d_in[4];
  const float* ln1_w = (const float*)d_in[5];
  const float* ln1_b = (const float*)d_in[6];
  const float* ln2_w = (const float*)d_in[7];
  const float* ln2_b = (const float*)d_in[8];
  const float* ff1_w = (const float*)d_in[9];
  const float* ff1_b = (const float*)d_in[10];
  const float* ff2_w = (const float*)d_in[11];
  const float* ff2_b = (const float*)d_in[12];

  bf16_t* w_qkv = (bf16_t*)d_ws;            // 288*96
  bf16_t* w_out = w_qkv + 288 * 96;         // 96*96
  bf16_t* w_ff1 = w_out + 96 * 96;          // 384*96
  bf16_t* w_ff2 = w_ff1 + 384 * 96;         // 96*384   total 221 KB bf16

  cvt_bf16<<<(288 * 96 + 255) / 256, 256, 0, stream>>>(qkv_w, w_qkv, 288 * 96);
  cvt_bf16<<<(96 * 96 + 255) / 256, 256, 0, stream>>>(out_w, w_out, 96 * 96);
  cvt_bf16<<<(384 * 96 + 255) / 256, 256, 0, stream>>>(ff1_w, w_ff1, 384 * 96);
  cvt_bf16<<<(96 * 384 + 255) / 256, 256, 0, stream>>>(ff2_w, w_ff2, 96 * 384);

  swin_block_kernel<<<128 * 64, 256, LDS_BYTES, stream>>>(
      x, w_qkv, qkv_b, w_out, out_b, ln1_w, ln1_b, ln2_w, ln2_b,
      w_ff1, ff1_b, w_ff2, ff2_b, (float*)d_out);
}
